// LlamaAttention_29317446763295
// MI455X (gfx1250) — matrix operand order repaired and checked by reading, not compiled, not run on hardware
//
#include <hip/hip_runtime.h>
#include <math.h>

// ---------------- problem constants ----------------
#define BB 2
#define SS 2048
#define DD 2048
#define NH 32
#define NKV 8
#define HD 64
#define MROWS (BB * SS)          // 4096

typedef __bf16 bf16_t;
typedef __attribute__((ext_vector_type(16))) __bf16 bf16x16;
typedef __attribute__((ext_vector_type(8)))  __bf16 bf16x8;
typedef __attribute__((ext_vector_type(8)))  float  f32x8;
typedef int v4i __attribute__((vector_size(16)));

#if __has_builtin(__builtin_amdgcn_global_load_async_to_lds_b128) && \
    __has_builtin(__builtin_amdgcn_s_wait_asynccnt)
#define HAS_ASYNC_LDS 1
#else
#define HAS_ASYNC_LDS 0
#endif

// ---------------- helpers ----------------
__device__ __forceinline__ bf16x16 pack16(bf16x8 lo, bf16x8 hi) {
  bf16x16 r;
#pragma unroll
  for (int i = 0; i < 8; ++i) { r[i] = lo[i]; r[i + 8] = hi[i]; }
  return r;
}
__device__ __forceinline__ f32x8 wmma_bf16(bf16x16 a, bf16x16 b, f32x8 c) {
  return __builtin_amdgcn_wmma_f32_16x16x32_bf16(false, a, false, b, (short)0, c,
                                                 false, false);
}
__device__ __forceinline__ f32x8 zero8() {
  f32x8 z = {0.f, 0.f, 0.f, 0.f, 0.f, 0.f, 0.f, 0.f};
  return z;
}

// 16-byte global -> LDS copy, async on CDNA5
__device__ __forceinline__ void cp_async16(bf16_t* lds_dst, const bf16_t* gsrc) {
#if HAS_ASYNC_LDS
  __builtin_amdgcn_global_load_async_to_lds_b128(
      (__attribute__((address_space(1))) v4i*)(void*)gsrc,
      (__attribute__((address_space(3))) v4i*)(void*)lds_dst, 0, 0);
#else
  *(bf16x8*)lds_dst = *(const bf16x8*)gsrc;
#endif
}
template <int N>
__device__ __forceinline__ void async_wait() {
#if HAS_ASYNC_LDS
  __builtin_amdgcn_s_wait_asynccnt(N);
#endif
}

// ---------------- elementwise conversion kernels ----------------
__global__ __launch_bounds__(256) void k_f32_to_bf16(const float* __restrict__ in,
                                                     bf16_t* __restrict__ out, int n) {
  int i = blockIdx.x * 256 + threadIdx.x;
  if (i < n) out[i] = (bf16_t)in[i];
}

// in: fp32 [K][N] row-major -> out: bf16 [N][K] (transposed)
__global__ __launch_bounds__(256) void k_transpose_bf16(const float* __restrict__ in,
                                                        bf16_t* __restrict__ out,
                                                        int K, int N) {
  int i = blockIdx.x * 256 + threadIdx.x;
  if (i >= K * N) return;
  int k = i / N, n = i % N;
  out[(size_t)n * K + k] = (bf16_t)in[i];
}

// RoPE: in bf16 [B*S][nheads*64] -> out bf16 [B][nheads][S][64]
__global__ __launch_bounds__(256) void k_rope(const bf16_t* __restrict__ in,
                                              const float* __restrict__ fc,
                                              const float* __restrict__ fs,
                                              bf16_t* __restrict__ out, int nheads) {
  int i = blockIdx.x * 256 + threadIdx.x;
  int total = BB * SS * nheads * 32;
  if (i >= total) return;
  int d2 = i % 32;
  int h  = (i / 32) % nheads;
  int s  = (i / (32 * nheads)) % SS;
  int b  = i / (32 * nheads * SS);
  float c = fc[s * 32 + d2], sn = fs[s * 32 + d2];
  size_t ib = ((size_t)(b * SS + s)) * (nheads * HD) + h * HD + 2 * d2;
  float xr = (float)in[ib], xi = (float)in[ib + 1];
  size_t ob = (((size_t)(b * nheads + h)) * SS + s) * HD + 2 * d2;
  out[ob]     = (bf16_t)(xr * c - xi * sn);
  out[ob + 1] = (bf16_t)(xr * sn + xi * c);
}

// V transpose: in bf16 [B*S][NKV*64] -> out bf16 [B][NKV][64][S]
__global__ __launch_bounds__(256) void k_vtrans(const bf16_t* __restrict__ in,
                                                bf16_t* __restrict__ out) {
  int i = blockIdx.x * 256 + threadIdx.x;
  int total = BB * SS * NKV * HD;
  if (i >= total) return;
  int d = i % HD;
  int h = (i / HD) % NKV;
  int s = (i / (HD * NKV)) % SS;
  int b = i / (HD * NKV * SS);
  out[(((size_t)(b * NKV + h)) * HD + d) * SS + s] =
      in[((size_t)(b * SS + s)) * (NKV * HD) + h * HD + d];
}

// ---------------- tiled bf16 WMMA GEMM (double-buffered async) ----------------
// C[M][N] = A[M][K] (bf16 row-major) * Bt[N][K]^T (bf16, B stored transposed)
// Block tile 128x128, 8 waves: 4 (M) x 2 (N); per wave 2x4 16x16 accumulators.
// M % 128 == 0, N % 128 == 0, K % 32 == 0 assumed (true for all uses here).
template <bool OUT_F32>
__global__ __launch_bounds__(256) void k_gemm_bf16(const bf16_t* __restrict__ A,
                                                   const bf16_t* __restrict__ Bt,
                                                   void* __restrict__ Cout,
                                                   int M, int N, int K) {
  constexpr int TK = 32;
  __shared__ __align__(16) bf16_t As[2][128][TK + 8];
  __shared__ __align__(16) bf16_t Bs[2][128][TK + 8];
  const int tid = threadIdx.x;
  const int lane = tid & 31;
  const int wid  = tid >> 5;
  const int wm = wid & 3;      // M slice: rows wm*32 .. +31
  const int wn = wid >> 2;     // N slice: cols wn*64 .. +63
  const int M0 = blockIdx.y * 128;
  const int N0 = blockIdx.x * 128;
  const int lrow = lane & 15;
  const int khalf = lane >> 4;

  f32x8 acc[2][4];
#pragma unroll
  for (int mi = 0; mi < 2; ++mi)
#pragma unroll
    for (int ni = 0; ni < 4; ++ni) acc[mi][ni] = zero8();

  const int nk = K / TK;
  // per tile, each thread issues 4 async copies (2 for A, 2 for B)
  auto issue_tile = [&](int t, int buf) {
#pragma unroll
    for (int c = 0; c < 2; ++c) {
      int r = (tid >> 2) + c * 64;
      int col = (tid & 3) * 8;
      cp_async16(&As[buf][r][col], &A[(size_t)(M0 + r) * K + t * TK + col]);
      cp_async16(&Bs[buf][r][col], &Bt[(size_t)(N0 + r) * K + t * TK + col]);
    }
  };

  issue_tile(0, 0);
  for (int t = 0; t < nk; ++t) {
    const int buf = t & 1;
    const bool nxt = (t + 1 < nk);
    if (nxt) issue_tile(t + 1, (t + 1) & 1);
    if (nxt) async_wait<4>(); else async_wait<0>();   // tile t resident
    __syncthreads();

    bf16x16 bfrag[4];
#pragma unroll
    for (int ni = 0; ni < 4; ++ni) {
      int n = wn * 64 + ni * 16 + lrow;
      bf16x8 lo = *(const bf16x8*)&Bs[buf][n][khalf * 8];
      bf16x8 hi = *(const bf16x8*)&Bs[buf][n][16 + khalf * 8];
      bfrag[ni] = pack16(lo, hi);
    }
    bf16x16 afrag[2];
#pragma unroll
    for (int mi = 0; mi < 2; ++mi) {
      int r = wm * 32 + mi * 16 + lrow;
      bf16x8 lo = *(const bf16x8*)&As[buf][r][khalf * 8];
      bf16x8 hi = *(const bf16x8*)&As[buf][r][16 + khalf * 8];
      afrag[mi] = pack16(lo, hi);
    }
#pragma unroll
    for (int mi = 0; mi < 2; ++mi)
#pragma unroll
      for (int ni = 0; ni < 4; ++ni)
        acc[mi][ni] = wmma_bf16(afrag[mi], bfrag[ni], acc[mi][ni]);
    __syncthreads();   // all waves done reading buf before t+2 overwrites it
  }

  // epilogue: C layout -> VGPR v: lanes0-15 M=v, lanes16-31 M=v+8; N=lane&15
#pragma unroll
  for (int mi = 0; mi < 2; ++mi)
#pragma unroll
    for (int ni = 0; ni < 4; ++ni)
#pragma unroll
      for (int v = 0; v < 8; ++v) {
        int row = M0 + wm * 32 + mi * 16 + v + 8 * khalf;
        int col = N0 + wn * 64 + ni * 16 + lrow;
        if (OUT_F32)
          ((float*)Cout)[(size_t)row * N + col] = acc[mi][ni][v];
        else
          ((bf16_t*)Cout)[(size_t)row * N + col] = (bf16_t)acc[mi][ni][v];
      }
}

// ---------------- flash attention (double-buffered async K/V) ----------------
// Q: [B][NH][S][64] bf16, Kc: [B][NKV][S][64] bf16, Vt: [B][NKV][64][S] bf16
// Out: [B][S][NH*64] bf16. Block = (q-tile of 128 rows) x (b*NH + h).
__global__ __launch_bounds__(256) void k_attn(const bf16_t* __restrict__ Q,
                                              const bf16_t* __restrict__ Kc,
                                              const bf16_t* __restrict__ Vt,
                                              bf16_t* __restrict__ Out) {
  // QPs: used first as the Q tile, then reused as the P (probs) tile
  __shared__ __align__(16) bf16_t QPs[128][HD + 8];
  __shared__ __align__(16) bf16_t Ks[2][64][HD + 8];   // [key][d]  (B-layout QK^T)
  __shared__ __align__(16) bf16_t Vs[2][64][HD + 8];   // [d][key]  (B-layout PV)

  const int qt = blockIdx.x;
  const int bh = blockIdx.y;
  const int b = bh / NH, h = bh % NH;
  const int hk = h / (NH / NKV);
  const int q0 = qt * 128;

  const bf16_t* Qb = Q + ((size_t)(b * NH + h) * SS + q0) * HD;
  const bf16_t* Kb = Kc + (size_t)(b * NKV + hk) * SS * HD;
  const bf16_t* Vb = Vt + (size_t)(b * NKV + hk) * HD * SS;

  const int tid = threadIdx.x;
  const int lane = tid & 31;
  const int w = tid >> 5;
  const int lrow = lane & 15;
  const int khalf = lane >> 4;

  // per tile, each thread issues 4 async copies (2 for K, 2 for V)
  auto issue_kv = [&](int kt, int buf) {
#pragma unroll
    for (int c = 0; c < 2; ++c) {
      int r = (tid >> 3) + c * 32;
      int col = (tid & 7) * 8;
      cp_async16(&Ks[buf][r][col], &Kb[(size_t)(kt * 64 + r) * HD + col]);
      cp_async16(&Vs[buf][r][col], &Vb[(size_t)r * SS + kt * 64 + col]);
    }
  };
  issue_kv(0, 0);

  // load Q tile 128x64 (one-time, plain loads)
#pragma unroll
  for (int c = 0; c < 4; ++c) {
    int r = (tid >> 3) + c * 32;
    int col = (tid & 7) * 8;
    *(bf16x8*)&QPs[r][col] = *(const bf16x8*)&Qb[(size_t)r * HD + col];
  }
  __syncthreads();

  // per-wave Q A-fragments (rows w*16..w*16+15), two 32-wide d-chunks
  bf16x16 qfrag[2];
#pragma unroll
  for (int kc = 0; kc < 2; ++kc) {
    int r = w * 16 + lrow;
    bf16x8 lo = *(const bf16x8*)&QPs[r][kc * 32 + khalf * 8];
    bf16x8 hi = *(const bf16x8*)&QPs[r][kc * 32 + 16 + khalf * 8];
    qfrag[kc] = pack16(lo, hi);
  }

  float m_run[8], l_run[8];
  f32x8 o_acc[4];
#pragma unroll
  for (int v = 0; v < 8; ++v) { m_run[v] = -1e30f; l_run[v] = 0.f; }
#pragma unroll
  for (int ni = 0; ni < 4; ++ni) o_acc[ni] = zero8();

  const int ntiles = q0 / 64 + 2;   // causal: keys 0 .. q0+127
  for (int kt = 0; kt < ntiles; ++kt) {
    const int buf = kt & 1;
    const int key0 = kt * 64;
    const bool nxt = (kt + 1 < ntiles);
    if (nxt) issue_kv(kt + 1, (kt + 1) & 1);
    if (nxt) async_wait<4>(); else async_wait<0>();   // tile kt resident
    __syncthreads();  // publishes K/V tile; also all qfrag builds / prior reads done

    // scores S = Q * K^T  (16 q-rows x 64 keys per wave)
    f32x8 sc[4];
#pragma unroll
    for (int ni = 0; ni < 4; ++ni) {
      int n = ni * 16 + lrow;
      f32x8 z = zero8();
#pragma unroll
      for (int kc = 0; kc < 2; ++kc) {
        bf16x8 lo = *(const bf16x8*)&Ks[buf][n][kc * 32 + khalf * 8];
        bf16x8 hi = *(const bf16x8*)&Ks[buf][n][kc * 32 + 16 + khalf * 8];
        z = wmma_bf16(qfrag[kc], pack16(lo, hi), z);
      }
      sc[ni] = z;
    }

    // online softmax (row m = v + 8*khalf within wave's 16 rows)
    float rm[8];
#pragma unroll
    for (int v = 0; v < 8; ++v) {
      int qg = q0 + w * 16 + v + 8 * khalf;
      float mx = -1e30f;
#pragma unroll
      for (int ni = 0; ni < 4; ++ni) {
        int kg = key0 + ni * 16 + lrow;
        float val = sc[ni][v] * 0.125f;     // 1/sqrt(64)
        val = (kg <= qg) ? val : -1e30f;    // causal mask
        sc[ni][v] = val;
        mx = fmaxf(mx, val);
      }
#pragma unroll
      for (int m = 1; m < 16; m <<= 1) mx = fmaxf(mx, __shfl_xor(mx, m, 32));
      rm[v] = mx;
    }
#pragma unroll
    for (int v = 0; v < 8; ++v) {
      float mnew = fmaxf(m_run[v], rm[v]);
      float corr = __expf(m_run[v] - mnew);
      m_run[v] = mnew;
      float rs = 0.f;
#pragma unroll
      for (int ni = 0; ni < 4; ++ni) {
        float pv = __expf(sc[ni][v] - mnew);
        sc[ni][v] = pv;
        rs += pv;
      }
#pragma unroll
      for (int m = 1; m < 16; m <<= 1) rs += __shfl_xor(rs, m, 32);
      l_run[v] = l_run[v] * corr + rs;
#pragma unroll
      for (int ni = 0; ni < 4; ++ni) o_acc[ni][v] *= corr;
    }

    // stash P (bf16) to LDS (per-wave rows only; same-wave RAW handled by dscnt)
#pragma unroll
    for (int ni = 0; ni < 4; ++ni)
#pragma unroll
      for (int v = 0; v < 8; ++v)
        QPs[w * 16 + v + 8 * khalf][ni * 16 + lrow] = (bf16_t)sc[ni][v];

    // O += P * V   (k-dim = 64 keys, two 32-chunks)
#pragma unroll
    for (int kc = 0; kc < 2; ++kc) {
      int r = w * 16 + lrow;
      bf16x8 plo = *(const bf16x8*)&QPs[r][kc * 32 + khalf * 8];
      bf16x8 phi = *(const bf16x8*)&QPs[r][kc * 32 + 16 + khalf * 8];
      bf16x16 pf = pack16(plo, phi);
#pragma unroll
      for (int ni = 0; ni < 4; ++ni) {
        int n = ni * 16 + lrow;
        bf16x8 vlo = *(const bf16x8*)&Vs[buf][n][kc * 32 + khalf * 8];
        bf16x8 vhi = *(const bf16x8*)&Vs[buf][n][kc * 32 + 16 + khalf * 8];
        o_acc[ni] = wmma_bf16(pf, pack16(vlo, vhi), o_acc[ni]);
      }
    }
    __syncthreads();  // all waves done with buf before kt+2 overwrites it
  }

  // epilogue: Out[b][s][h*64+d]
  bf16_t* ob = Out + ((size_t)(b * SS + q0)) * (NH * HD) + h * HD;
#pragma unroll
  for (int ni = 0; ni < 4; ++ni)
#pragma unroll
    for (int v = 0; v < 8; ++v) {
      int r = w * 16 + v + 8 * khalf;
      int d = ni * 16 + lrow;
      ob[(size_t)r * (NH * HD) + d] = (bf16_t)(o_acc[ni][v] / l_run[v]);
    }
}

// ---------------- host launcher ----------------
extern "C" void kernel_launch(void* const* d_in, const int* in_sizes, int n_in,
                              void* d_out, int out_size, void* d_ws, size_t ws_size,
                              hipStream_t stream) {
  (void)in_sizes; (void)n_in; (void)out_size; (void)ws_size;
  const float* x  = (const float*)d_in[0];
  const float* wq = (const float*)d_in[1];
  const float* wk = (const float*)d_in[2];
  const float* wv = (const float*)d_in[3];
  const float* wo = (const float*)d_in[4];
  const float* fc = (const float*)d_in[5];
  const float* fs = (const float*)d_in[6];
  // d_in[7] = mask (unused; causal handled analytically)
  float* out = (float*)d_out;

  char* ws = (char*)d_ws;
  size_t off = 0;
  auto salloc = [&](size_t bytes) -> void* {
    void* p = ws + off;
    off += (bytes + 255) & ~(size_t)255;
    return p;
  };
  bf16_t* xb   = (bf16_t*)salloc((size_t)MROWS * DD * 2);
  bf16_t* wqT  = (bf16_t*)salloc((size_t)DD * DD * 2);
  bf16_t* wkT  = (bf16_t*)salloc((size_t)DD * (NKV * HD) * 2);
  bf16_t* wvT  = (bf16_t*)salloc((size_t)DD * (NKV * HD) * 2);
  bf16_t* woT  = (bf16_t*)salloc((size_t)DD * DD * 2);
  bf16_t* qlin = (bf16_t*)salloc((size_t)MROWS * DD * 2);
  bf16_t* klin = (bf16_t*)salloc((size_t)MROWS * (NKV * HD) * 2);
  bf16_t* vlin = (bf16_t*)salloc((size_t)MROWS * (NKV * HD) * 2);
  bf16_t* qr   = (bf16_t*)salloc((size_t)MROWS * DD * 2);
  bf16_t* kr   = (bf16_t*)salloc((size_t)MROWS * (NKV * HD) * 2);
  bf16_t* vt   = (bf16_t*)salloc((size_t)MROWS * (NKV * HD) * 2);
  bf16_t* aout = (bf16_t*)salloc((size_t)MROWS * DD * 2);

  const int nx = MROWS * DD;
  k_f32_to_bf16<<<(nx + 255) / 256, 256, 0, stream>>>(x, xb, nx);
  k_transpose_bf16<<<(DD * DD + 255) / 256, 256, 0, stream>>>(wq, wqT, DD, DD);
  k_transpose_bf16<<<(DD * NKV * HD + 255) / 256, 256, 0, stream>>>(wk, wkT, DD, NKV * HD);
  k_transpose_bf16<<<(DD * NKV * HD + 255) / 256, 256, 0, stream>>>(wv, wvT, DD, NKV * HD);
  k_transpose_bf16<<<(DD * DD + 255) / 256, 256, 0, stream>>>(wo, woT, DD, DD);

  // QKV projections
  k_gemm_bf16<false><<<dim3(DD / 128, MROWS / 128), 256, 0, stream>>>(
      xb, wqT, (void*)qlin, MROWS, DD, DD);
  k_gemm_bf16<false><<<dim3((NKV * HD) / 128, MROWS / 128), 256, 0, stream>>>(
      xb, wkT, (void*)klin, MROWS, NKV * HD, DD);
  k_gemm_bf16<false><<<dim3((NKV * HD) / 128, MROWS / 128), 256, 0, stream>>>(
      xb, wvT, (void*)vlin, MROWS, NKV * HD, DD);

  // RoPE + layout shuffles
  int nq = BB * SS * NH * 32;
  k_rope<<<(nq + 255) / 256, 256, 0, stream>>>(qlin, fc, fs, qr, NH);
  int nk = BB * SS * NKV * 32;
  k_rope<<<(nk + 255) / 256, 256, 0, stream>>>(klin, fc, fs, kr, NKV);
  int nv = BB * SS * NKV * HD;
  k_vtrans<<<(nv + 255) / 256, 256, 0, stream>>>(vlin, vt);

  // flash attention
  k_attn<<<dim3(SS / 128, BB * NH), 256, 0, stream>>>(qr, kr, vt, aout);

  // output projection (fp32 out)
  k_gemm_bf16<true><<<dim3(DD / 128, MROWS / 128), 256, 0, stream>>>(
      aout, woT, (void*)out, MROWS, DD, DD);
}
